// CMAttention_49246095016450
// MI455X (gfx1250) — compile-verified, matrix operand order checked by reading, not run on hardware
//
#include <hip/hip_runtime.h>

typedef unsigned short u16;
typedef __attribute__((ext_vector_type(16))) __bf16 v16bf;
typedef __attribute__((ext_vector_type(8)))  float  v8f;

#define NPADC 2112   // cross keys padded (2051 valid)
#define NVALC 2051
#define NPADS 1152   // self keys padded (1027 valid), multiple of 128
#define NVALS 1027
#define ATTN_SCALE 0.125f   // 64^-0.5

// ---------------- bf16 helpers ----------------
__device__ __forceinline__ u16 f2bf(float x) {
  union { float f; unsigned u; } c; c.f = x;
  unsigned r = c.u + 0x7FFFu + ((c.u >> 16) & 1u);
  return (u16)(r >> 16);
}
__device__ __forceinline__ float bf2f(u16 h) {
  union { unsigned u; float f; } c; c.u = ((unsigned)h) << 16;
  return c.f;
}

// Per-lane WMMA 16x16x32 bf16 fragment load from a row-major [rows x K] matrix.
// base points at element (tile_row0, k0); per ISA 7.12.2: lane L holds row (L&15),
// K-runs [kh*8, kh*8+8) and [16+kh*8, 16+kh*8+8) with kh = L>>4.
__device__ __forceinline__ v16bf load_frag(const u16* __restrict__ base, int ld) {
  int lane = threadIdx.x & 31;
  const u16* p = base + (lane & 15) * ld + ((lane >> 4) << 3);
  union { v16bf v; float4 q[2]; } u;
  u.q[0] = *reinterpret_cast<const float4*>(p);
  u.q[1] = *reinterpret_cast<const float4*>(p + 16);
  return u.v;
}
__device__ __forceinline__ v8f wmma_bf16(v16bf a, v16bf b, v8f c) {
  return __builtin_amdgcn_wmma_f32_16x16x32_bf16(false, a, false, b, (short)0, c, false, false);
}

// ---------------- conversion / layout kernels ----------------
__global__ void cvt_bf16(const float* __restrict__ in, u16* __restrict__ out, int count) {
  int i = blockIdx.x * blockDim.x + threadIdx.x;
  if (i < count) out[i] = f2bf(in[i]);
}
__global__ void build_ctx(const float* __restrict__ x, const float* __restrict__ y,
                          u16* __restrict__ out) {
  int i = blockIdx.x * blockDim.x + threadIdx.x;   // 16384*512
  if (i >= 16384 * 512) return;
  int row = i >> 9, c = i & 511;
  int b = row >> 11, t = row & 2047;
  float v = (t < 1024) ? x[((size_t)(b * 1024 + t) << 9) + c]
                       : y[((size_t)(b * 1024 + t - 1024) << 9) + c];
  out[i] = f2bf(v);
}
__global__ void build_wf(const float* __restrict__ Wf, u16* __restrict__ out) {
  int i = blockIdx.x * blockDim.x + threadIdx.x;   // NPADS*NPADC
  if (i >= NPADS * NPADC) return;
  int r = i / NPADC, c = i - r * NPADC;
  float v = (r < 1027 && c < 2051) ? Wf[(size_t)r * 2051 + c] : 0.f;
  out[i] = f2bf(v);
}
// mem tokens: value at (h,j,d) = scale * mem[flat>>9][flat&511], flat = h*192+j*64+d
__global__ void fill_mem_rows(const float* __restrict__ mem, u16* __restrict__ dst,
                              int nPad, int rowStart, int nFill, float scale) {
  int i = blockIdx.x * blockDim.x + threadIdx.x;   // 64 bh * nFill rows * 64 d
  if (i >= 64 * nFill * 64) return;
  int d = i & 63, t = i >> 6;
  int j2 = t % nFill, bh = t / nFill;
  int h = bh & 7;
  float v = 0.f;
  if (j2 < 3) {
    int flat = h * 192 + j2 * 64 + d;
    v = scale * mem[(flat >> 9) * 512 + (flat & 511)];
  }
  dst[((size_t)bh * nPad + rowStart + j2) * 64 + d] = f2bf(v);
}
__global__ void fill_mem_svt(const float* __restrict__ mem, u16* __restrict__ dst,
                             int nFill, float scale) {
  int i = blockIdx.x * blockDim.x + threadIdx.x;   // 64 bh * 64 d * nFill j
  if (i >= 64 * 64 * nFill) return;
  int j2 = i % nFill;
  int t = i / nFill;
  int d = t & 63, bh = t >> 6;
  int h = bh & 7;
  float v = 0.f;
  if (j2 < 3) {
    int flat = h * 192 + j2 * 64 + d;
    v = scale * mem[(flat >> 9) * 512 + (flat & 511)];
  }
  dst[((size_t)bh * 64 + d) * NPADS + 1024 + j2] = f2bf(v);
}

// ---------------- generic WMMA GEMM  C = A(MxK) * B(NxK)^T ----------------
// 8 waves: 4 row-waves x 2 col-waves. Wave tile = 32 rows x (NFRAG*16) cols.
// WG tile = 128 rows x (NFRAG*32) cols.
enum { M_QBF = 0, M_KCTX = 1, M_SKV = 2, M_ATT = 3, M_OH = 4, M_OUT = 5 };

template <int MODE, int NFRAG>
__global__ __launch_bounds__(256)
void gemm_bf16(const u16* __restrict__ Aall, const u16* __restrict__ Ball,
               const float* __restrict__ bias,
               u16* __restrict__ OutA, u16* __restrict__ OutB, float* __restrict__ OutF,
               int M, int N, int K, int lda, int ldb) {
  int z = blockIdx.z;
  const u16* A = Aall;
  const u16* B = Ball;
  if (MODE == M_ATT) { A = Aall + (size_t)z * 1024 * NPADC; }
  if (MODE == M_OH)  { A = Aall + (size_t)z * 1024 * NPADS; B = Ball + (size_t)z * 64 * NPADS; }

  int w = threadIdx.x >> 5;
  int lane = threadIdx.x & 31;
  int rowBase = blockIdx.y * 128 + (w >> 1) * 32;
  int colBase = blockIdx.x * (NFRAG * 32) + (w & 1) * (NFRAG * 16);

  v8f acc[2][NFRAG];
#pragma unroll
  for (int fr = 0; fr < 2; ++fr)
#pragma unroll
    for (int f = 0; f < NFRAG; ++f) acc[fr][f] = v8f{};

  const u16* Ar0 = A + (size_t)rowBase * lda;
  const u16* Ar1 = A + (size_t)(rowBase + 16) * lda;
  const u16* Bc[NFRAG];
#pragma unroll
  for (int f = 0; f < NFRAG; ++f) Bc[f] = B + (size_t)(colBase + f * 16) * ldb;

  for (int kb = 0; kb < K; kb += 32) {
    v16bf a0 = load_frag(Ar0 + kb, lda);
    v16bf a1 = load_frag(Ar1 + kb, lda);
    v16bf bfr[NFRAG];
#pragma unroll
    for (int f = 0; f < NFRAG; ++f) bfr[f] = load_frag(Bc[f] + kb, ldb);
#pragma unroll
    for (int f = 0; f < NFRAG; ++f) {
      acc[0][f] = wmma_bf16(a0, bfr[f], acc[0][f]);
      acc[1][f] = wmma_bf16(a1, bfr[f], acc[1][f]);
    }
  }

  int n = lane & 15;
  int hi = lane >> 4;
  auto emit = [&](int row, int col, float val) {
    if (MODE == M_QBF) {
      val += bias[col];
      OutA[(size_t)row * 512 + col] = f2bf(val);
    } else if (MODE == M_KCTX) {
      val += bias[col];
      int b = row >> 11, j = row & 2047;
      int h = col >> 6, d = col & 63;
      OutA[((size_t)(b * 8 + h) * NPADC + j) * 64 + d] = f2bf(val);
    } else if (MODE == M_SKV) {
      val += bias[col];
      int b = row >> 10, i = row & 1023;
      if (col < 512) {
        int h = col >> 6, d = col & 63;
        OutA[((size_t)(b * 8 + h) * NPADS + i) * 64 + d] = f2bf(val);
      } else {
        int c2 = col - 512;
        int h = c2 >> 6, d = c2 & 63;
        OutB[((size_t)(b * 8 + h) * 64 + d) * NPADS + i] = f2bf(val);
      }
    } else if (MODE == M_ATT) {
      if (col < NVALS) val += bias[col];
      size_t idx = (size_t)z * 1024 * NPADS + (size_t)row * NPADS + col;
      val += bf2f(OutA[idx]);        // add self_attn, write back in place
      OutA[idx] = f2bf(val);
    } else if (MODE == M_OH) {
      int b = z >> 3, h = z & 7;
      OutA[(size_t)(b * 1024 + row) * 512 + h * 64 + col] = f2bf(val);
    } else {                          // M_OUT
      val += bias[col];
      OutF[(size_t)row * 512 + col] = val;
    }
  };
#pragma unroll
  for (int fr = 0; fr < 2; ++fr)
#pragma unroll
    for (int f = 0; f < NFRAG; ++f)
#pragma unroll
      for (int i = 0; i < 8; ++i)
        emit(rowBase + fr * 16 + hi * 8 + i, colBase + f * 16 + n, acc[fr][f][i]);
}

// ---------------- fused scores (q.K^T * scale) + softmax -> bf16 probs ----------------
// grid: (64 i-tiles, 64 bh). 16 q-rows per WG; full score strip lives in dynamic LDS.
// Each wave processes two 16-col chunks at a time (independent WMMA chains).
__global__ __launch_bounds__(256)
void scores_softmax(const u16* __restrict__ qbf, const u16* __restrict__ Kmat,
                    u16* __restrict__ Out, int nValid, int nPad) {
  extern __shared__ float smem[];
  const int S = nPad + 20;                 // bank-stagger
  int bh = blockIdx.y;
  int it = blockIdx.x;
  int b = bh >> 3, h = bh & 7;
  const u16* qt = qbf + (size_t)(b * 1024 + it * 16) * 512 + h * 64;
  const u16* Kb = Kmat + (size_t)bh * nPad * 64;
  u16* Ob = Out + ((size_t)bh * 1024 + it * 16) * (size_t)nPad;

  int w = threadIdx.x >> 5;
  int lane = threadIdx.x & 31;
  int n = lane & 15, hi = lane >> 4;

  v16bf a0 = load_frag(qt, 512);           // q tile, K=0..31
  v16bf a1 = load_frag(qt + 32, 512);      // q tile, K=32..63
  int nch = nPad >> 4;                     // multiple of 2 for all our pads
  for (int c = w * 2; c < nch; c += 16) {
    v16bf b00 = load_frag(Kb + (size_t)c * 16 * 64, 64);
    v16bf b01 = load_frag(Kb + (size_t)c * 16 * 64 + 32, 64);
    v16bf b10 = load_frag(Kb + (size_t)(c + 1) * 16 * 64, 64);
    v16bf b11 = load_frag(Kb + (size_t)(c + 1) * 16 * 64 + 32, 64);
    v8f acc0{}, acc1{};
    acc0 = wmma_bf16(a0, b00, acc0);
    acc1 = wmma_bf16(a0, b10, acc1);
    acc0 = wmma_bf16(a1, b01, acc0);
    acc1 = wmma_bf16(a1, b11, acc1);
    int col0 = c * 16 + n;
#pragma unroll
    for (int i = 0; i < 8; ++i) {
      smem[(i + 8 * hi) * S + col0] = acc0[i] * ATTN_SCALE;
      smem[(i + 8 * hi) * S + col0 + 16] = acc1[i] * ATTN_SCALE;
    }
  }
  __syncthreads();

  // 16 lanes per row; rows 2w,2w+1 live in wave w -> width-16 shuffle reduce
  int r = threadIdx.x >> 4;
  int l16 = threadIdx.x & 15;
  float* row = smem + r * S;
  float mx = -3.0e38f;
  for (int j = l16; j < nValid; j += 16) mx = fmaxf(mx, row[j]);
  for (int off = 8; off > 0; off >>= 1) mx = fmaxf(mx, __shfl_xor(mx, off, 16));
  float sum = 0.f;
  for (int j = l16; j < nValid; j += 16) {
    float e = __expf(row[j] - mx);
    row[j] = e;
    sum += e;
  }
  for (int off = 8; off > 0; off >>= 1) sum += __shfl_xor(sum, off, 16);
  float* invs = smem + 16 * S;
  if (l16 == 0) invs[r] = 1.0f / sum;
  __syncthreads();

  for (int f = threadIdx.x; f < 16 * nPad; f += 256) {
    int rr = f / nPad, j = f - rr * nPad;
    float v = (j < nValid) ? smem[rr * S + j] * invs[rr] : 0.f;
    Ob[(size_t)rr * nPad + j] = f2bf(v);
  }
}

// ---------------- launch ----------------
extern "C" void kernel_launch(void* const* d_in, const int* in_sizes, int n_in,
                              void* d_out, int out_size, void* d_ws, size_t ws_size,
                              hipStream_t stream) {
  const float* inp = (const float*)d_in[0];
  const float* x   = (const float*)d_in[1];
  const float* y   = (const float*)d_in[2];
  const float* Wq  = (const float*)d_in[3];
  const float* bq  = (const float*)d_in[4];
  const float* Wkv = (const float*)d_in[5];
  const float* bkv = (const float*)d_in[6];
  const float* Wf  = (const float*)d_in[7];
  const float* bfv = (const float*)d_in[8];
  const float* Wo  = (const float*)d_in[9];
  const float* bo  = (const float*)d_in[10];
  const float* m_k = (const float*)d_in[11];
  // d_in[12] = m_v : unused by the reference output
  const float* Sm_k = (const float*)d_in[13];
  const float* Sm_v = (const float*)d_in[14];
  float* out = (float*)d_out;

  char* ws = (char*)d_ws;
  size_t off = 0;
  auto alloc = [&](size_t bytes) -> void* {
    void* p = ws + off;
    off = (off + bytes + 255) & ~(size_t)255;
    return p;
  };
  u16* inp_bf = (u16*)alloc(8192ull * 512 * 2);
  u16* ctx_bf = (u16*)alloc(16384ull * 512 * 2);
  u16* Wq_bf  = (u16*)alloc(512ull * 512 * 2);
  u16* Wkv_bf = (u16*)alloc(1024ull * 512 * 2);
  u16* Wf_bf  = (u16*)alloc((size_t)NPADS * NPADC * 2);
  u16* Wo_bf  = (u16*)alloc(512ull * 512 * 2);
  u16* q_bf   = (u16*)alloc(8192ull * 512 * 2);
  u16* Kpad   = (u16*)alloc(64ull * NPADC * 64 * 2);
  u16* Skpad  = (u16*)alloc(64ull * NPADS * 64 * 2);
  u16* SvT    = (u16*)alloc(64ull * 64 * NPADS * 2);
  u16* CA     = (u16*)alloc(64ull * 1024 * NPADC * 2);
  u16* SA     = (u16*)alloc(64ull * 1024 * NPADS * 2);   // later reused as ATT (in place)
  u16* OH     = (u16*)alloc(8192ull * 512 * 2);

  const int T = 256;
  // stage 0: bf16 conversions / layout builds
  cvt_bf16<<<(4194304 + T - 1) / T, T, 0, stream>>>(inp, inp_bf, 4194304);
  cvt_bf16<<<(262144 + T - 1) / T, T, 0, stream>>>(Wq, Wq_bf, 262144);
  cvt_bf16<<<(524288 + T - 1) / T, T, 0, stream>>>(Wkv, Wkv_bf, 524288);
  cvt_bf16<<<(262144 + T - 1) / T, T, 0, stream>>>(Wo, Wo_bf, 262144);
  build_ctx<<<(16384 * 512 + T - 1) / T, T, 0, stream>>>(x, y, ctx_bf);
  build_wf<<<(NPADS * NPADC + T - 1) / T, T, 0, stream>>>(Wf, Wf_bf);

  // stage 1: projections (WMMA). q = inp@Wq.T ; Kctx = ctx@Wkv[:512].T ; skv = inp@Wkv.T
  gemm_bf16<M_QBF, 4><<<dim3(4, 64, 1), T, 0, stream>>>(inp_bf, Wq_bf, bq, q_bf, nullptr, nullptr,
                                                        8192, 512, 512, 512, 512);
  gemm_bf16<M_KCTX, 4><<<dim3(4, 128, 1), T, 0, stream>>>(ctx_bf, Wkv_bf, bkv, Kpad, nullptr,
                                                          nullptr, 16384, 512, 512, 512, 512);
  gemm_bf16<M_SKV, 4><<<dim3(8, 64, 1), T, 0, stream>>>(inp_bf, Wkv_bf, bkv, Skpad, SvT, nullptr,
                                                        8192, 1024, 512, 512, 512);
  // memory tokens (+ zero pad rows/cols)
  fill_mem_rows<<<(64 * 64 * 64 + T - 1) / T, T, 0, stream>>>(m_k, Kpad, NPADC, 2048, 64, 8.0f);
  fill_mem_rows<<<(64 * 128 * 64 + T - 1) / T, T, 0, stream>>>(Sm_k, Skpad, NPADS, 1024, 128, 8.0f);
  fill_mem_svt<<<(64 * 64 * 128 + T - 1) / T, T, 0, stream>>>(Sm_v, SvT, 128, 1.73205081f);

  // stage 2: fused scores + softmax (dynamic LDS holds the full 16-row score strip)
  size_t smemC = (size_t)(16 * (NPADC + 20) + 16) * sizeof(float);   // ~136.5 KB
  size_t smemS = (size_t)(16 * (NPADS + 20) + 16) * sizeof(float);   // ~75 KB
  scores_softmax<<<dim3(64, 64), T, smemC, stream>>>(q_bf, Kpad, CA, NVALC, NPADC);
  scores_softmax<<<dim3(64, 64), T, smemS, stream>>>(q_bf, Skpad, SA, NVALS, NPADS);

  // stage 3: big Wf GEMM (+bias +self_attn, in place over SA) -> ATT
  gemm_bf16<M_ATT, 4><<<dim3(NPADS / 128, 8, 64), T, 0, stream>>>(CA, Wf_bf, bfv, SA, nullptr,
                                                                  nullptr, 1024, NPADS, NPADC,
                                                                  NPADC, NPADC);
  // stage 4: attn @ Sv -> per-head output (narrow N=64 -> 2-fragment config)
  gemm_bf16<M_OH, 2><<<dim3(1, 8, 64), T, 0, stream>>>(SA, SvT, nullptr, OH, nullptr, nullptr,
                                                       1024, 64, NPADS, NPADS, NPADS);
  // stage 5: final projection -> fp32 output
  gemm_bf16<M_OUT, 4><<<dim3(4, 64, 1), T, 0, stream>>>(OH, Wo_bf, bo, nullptr, nullptr, out,
                                                        8192, 512, 512, 512, 512);
}
